// Head_62466004353686
// MI455X (gfx1250) — matrix operand order repaired and checked by reading, not compiled, not run on hardware
//
#include <hip/hip_runtime.h>
#include <hip/hip_bf16.h>
#include <cstdint>

typedef __attribute__((ext_vector_type(16))) _Float16 v16h;
typedef __attribute__((ext_vector_type(8)))  _Float16 v8h;
typedef __attribute__((ext_vector_type(8)))  float    v8f;
typedef __attribute__((ext_vector_type(4)))  float    v4f;

#define B_DIM 4
#define T_SEQ 4096
#define C_DIM 1024
#define H_DIM 64

union U16h { v16h v; v8h h[2]; };

// ---------------------------------------------------------------------------
// Kernel 0: pack Wq/Wk/Wv (f32 row-major) into f16 B-fragment order, grouped
// so each k-chunk's 12 fragments (3 mats x 4 n-tiles) are contiguous 12KB.
// pw[((kc*12 + m*4 + n)*32 + lane)*16 + i]
// ---------------------------------------------------------------------------
__global__ __launch_bounds__(256) void pack_w_kernel(
    const float* __restrict__ Wq, const float* __restrict__ Wk,
    const float* __restrict__ Wv, _Float16* __restrict__ pw)
{
    int tid  = blockIdx.x * blockDim.x + threadIdx.x;   // 0..12287
    int lane = tid & 31;
    int n    = (tid >> 5) & 3;
    int kc   = (tid >> 7) & 31;
    int m    = tid >> 12;                               // 0=q 1=k 2=v
    const float* W = (m == 0) ? Wq : (m == 1) ? Wk : Wv;
    int l15 = lane & 15, grp = lane >> 4;
    _Float16* dst = pw + (((long)kc * 12 + m * 4 + n) * 32 + lane) * 16;
    #pragma unroll
    for (int i = 0; i < 16; ++i)
        dst[i] = (_Float16)W[(long)(kc * 32 + grp * 8 + (i & 7) + ((i >> 3) << 4)) * H_DIM + n * 16 + l15];
}

// ---------------------------------------------------------------------------
// Async prefetch of one 12KB W chunk into LDS; each of 8 waves moves 3x512B.
// ---------------------------------------------------------------------------
__device__ __forceinline__ void prefetch_w(const _Float16* src, unsigned ldsDst,
                                           int wslot, int lane)
{
    unsigned off0 = (unsigned)(wslot * 1536 + lane * 16);
    unsigned vd   = ldsDst + off0;
    uint64_t sb   = (uint64_t)(uintptr_t)src;
    asm volatile(
        "global_load_async_to_lds_b128 %0, %1, %2 offset:0\n\t"
        "global_load_async_to_lds_b128 %0, %1, %2 offset:512\n\t"
        "global_load_async_to_lds_b128 %0, %1, %2 offset:1024"
        :
        : "v"(vd), "v"(off0), "s"(sb)
        : "memory");
}

// ---------------------------------------------------------------------------
// Kernel 1: q/k/v = x @ W. W chunk double-buffered in LDS (block-wide async
// copy), x A-fragment software-pipelined in registers.
// ---------------------------------------------------------------------------
__global__ __launch_bounds__(256) void qkv_proj_kernel(
    const float* __restrict__ x,
    const _Float16* __restrict__ pw,
    _Float16* __restrict__ qh,
    _Float16* __restrict__ kh,
    _Float16* __restrict__ vpk)
{
    __shared__ __align__(16) _Float16 wbuf[2][12 * 512];   // 2 x 12KB

    const int lane  = threadIdx.x & 31;
    const int wslot = threadIdx.x >> 5;
    const int wid   = (int)((blockIdx.x * blockDim.x + threadIdx.x) >> 5); // 0..1023
    const int l15   = lane & 15;
    const int grp   = lane >> 4;
    const long rowBase = (long)wid * 16;
    const unsigned wbase = (unsigned)(uintptr_t)&wbuf[0][0];

    v8f acc[12];
    #pragma unroll
    for (int f = 0; f < 12; ++f) acc[f] = v8f{};

    const float* xp0 = x + (rowBase + l15) * C_DIM + (grp << 3);

    // prologue: W chunk 0 into buf 0; first A-fragment raw loads
    prefetch_w(pw, wbase, wslot, lane);
    v4f xa0 = *(const v4f*)(xp0);
    v4f xa1 = *(const v4f*)(xp0 + 4);
    v4f xa2 = *(const v4f*)(xp0 + 16);
    v4f xa3 = *(const v4f*)(xp0 + 20);
    asm volatile("s_wait_asynccnt 0" ::: "memory");
    __syncthreads();

    for (int kc = 0; kc < 32; ++kc) {
        const int cur = kc & 1;
        v4f xn0, xn1, xn2, xn3;
        if (kc + 1 < 32) {
            // prefetch next W chunk (12KB) and next x A-fragment
            prefetch_w(pw + (long)(kc + 1) * 6144,
                       wbase + (cur ^ 1) * 12288, wslot, lane);
            const float* xp = xp0 + (kc + 1) * 32;
            xn0 = *(const v4f*)(xp);
            xn1 = *(const v4f*)(xp + 4);
            xn2 = *(const v4f*)(xp + 16);
            xn3 = *(const v4f*)(xp + 20);
        }

        v16h a;
        #pragma unroll
        for (int i = 0; i < 4; ++i) {
            a[i]      = (_Float16)xa0[i];
            a[4 + i]  = (_Float16)xa1[i];
            a[8 + i]  = (_Float16)xa2[i];
            a[12 + i] = (_Float16)xa3[i];
        }

        #pragma unroll
        for (int f = 0; f < 12; ++f) {
            const _Float16* fp = &wbuf[cur][f * 512 + lane * 16];
            U16h bf;
            bf.h[0] = *(const v8h*)(fp);        // ds_load_b128
            bf.h[1] = *(const v8h*)(fp + 8);
            acc[f] = __builtin_amdgcn_wmma_f32_16x16x32_f16(
                false, a, false, bf.v, (short)0, acc[f], false, false);
        }

        if (kc + 1 < 32) {
            xa0 = xn0; xa1 = xn1; xa2 = xn2; xa3 = xn3;
            asm volatile("s_wait_asynccnt 0" ::: "memory");
            __syncthreads();                     // next buffer ready, cur reads done
        }
    }

    // q,k row-major (C layout: M = r + 8*grp, N = l15); q pre-scaled 1/8
    _Float16* outRM[2] = { qh, kh };
    #pragma unroll
    for (int m = 0; m < 2; ++m) {
        const float scl = (m == 0) ? 0.125f : 1.0f;
        #pragma unroll
        for (int n = 0; n < 4; ++n)
            #pragma unroll
            for (int r = 0; r < 8; ++r) {
                long row = rowBase + r + (grp << 3);
                outRM[m][row * H_DIM + n * 16 + l15] = (_Float16)(acc[m * 4 + n][r] * scl);
            }
    }
    // v in packed B-fragment order
    #pragma unroll
    for (int n = 0; n < 4; ++n)
        #pragma unroll
        for (int r = 0; r < 8; ++r) {
            int s = (int)rowBase + r + (grp << 3);
            long addr = ((((long)(s >> 5) * 4 + n) * 32) + ((s >> 4) & 1) * 16 + l15) * 16 + (s & 15);
            vpk[addr] = (_Float16)acc[8 + n][r];
        }
}

// ---------------------------------------------------------------------------
// Async prefetch of one 4KB K tile + 4KB packed-V tile into LDS (16 ops).
// ---------------------------------------------------------------------------
__device__ __forceinline__ void prefetch_tile(const _Float16* ksrc, const _Float16* vsrc,
                                              unsigned kdst, unsigned vdst, int lane)
{
    unsigned voff = (unsigned)(lane * 16);
    unsigned kd = kdst + voff;
    unsigned vd = vdst + voff;
    uint64_t kb = (uint64_t)(uintptr_t)ksrc;
    uint64_t vb = (uint64_t)(uintptr_t)vsrc;
    asm volatile(
        "global_load_async_to_lds_b128 %0, %2, %3 offset:0\n\t"
        "global_load_async_to_lds_b128 %0, %2, %3 offset:512\n\t"
        "global_load_async_to_lds_b128 %0, %2, %3 offset:1024\n\t"
        "global_load_async_to_lds_b128 %0, %2, %3 offset:1536\n\t"
        "global_load_async_to_lds_b128 %0, %2, %3 offset:2048\n\t"
        "global_load_async_to_lds_b128 %0, %2, %3 offset:2560\n\t"
        "global_load_async_to_lds_b128 %0, %2, %3 offset:3072\n\t"
        "global_load_async_to_lds_b128 %0, %2, %3 offset:3584\n\t"
        "global_load_async_to_lds_b128 %1, %2, %4 offset:0\n\t"
        "global_load_async_to_lds_b128 %1, %2, %4 offset:512\n\t"
        "global_load_async_to_lds_b128 %1, %2, %4 offset:1024\n\t"
        "global_load_async_to_lds_b128 %1, %2, %4 offset:1536\n\t"
        "global_load_async_to_lds_b128 %1, %2, %4 offset:2048\n\t"
        "global_load_async_to_lds_b128 %1, %2, %4 offset:2560\n\t"
        "global_load_async_to_lds_b128 %1, %2, %4 offset:3072\n\t"
        "global_load_async_to_lds_b128 %1, %2, %4 offset:3584"
        :
        : "v"(kd), "v"(vd), "v"(voff), "s"(kb), "s"(vb)
        : "memory");
}

// ---------------------------------------------------------------------------
// Kernel 2: causal flash attention, 32 query rows per wave, K/V double-
// buffered in LDS via async-to-LDS copies (ASYNCcnt). 4 waves/block, 72KB LDS.
// ---------------------------------------------------------------------------
__global__ __launch_bounds__(128) void attn_kernel(
    const _Float16* __restrict__ qh,
    const _Float16* __restrict__ kh,
    const _Float16* __restrict__ vpk,
    float* __restrict__ out)
{
    __shared__ __align__(16) _Float16 kbuf[4][2][32 * 64];
    __shared__ __align__(16) _Float16 vbuf[4][2][32 * 64];
    __shared__ __align__(16) _Float16 plds[4][2][16 * 32];

    const int lane  = threadIdx.x & 31;
    const int wslot = threadIdx.x >> 5;
    const int wid   = (int)((blockIdx.x * blockDim.x + threadIdx.x) >> 5); // 0..511
    const int l15   = lane & 15;
    const int grp   = lane >> 4;
    const int b     = wid >> 7;               // 128 query tiles (of 32 rows) per batch
    const int t0    = (wid & 127) << 5;
    const long bT   = (long)b * T_SEQ;

    const unsigned kbase = (unsigned)(uintptr_t)&kbuf[wslot][0][0];
    const unsigned vbase = (unsigned)(uintptr_t)&vbuf[wslot][0][0];

    // Q: two row tiles x two K-chunks of A fragments (pre-scaled by 1/8)
    v16h aq[2][2];
    #pragma unroll
    for (int rt = 0; rt < 2; ++rt) {
        const long qrow = (bT + t0 + rt * 16 + l15) * H_DIM;
        #pragma unroll
        for (int hc = 0; hc < 2; ++hc) {
            const _Float16* qp = qh + qrow + hc * 32 + (grp << 3);
            U16h u;
            u.h[0] = *(const v8h*)(qp);
            u.h[1] = *(const v8h*)(qp + 16);
            aq[rt][hc] = u.v;
        }
    }

    v8f o[2][4];
    float mrow[2][8], lrow[2][8];
    #pragma unroll
    for (int rt = 0; rt < 2; ++rt) {
        #pragma unroll
        for (int n = 0; n < 4; ++n) o[rt][n] = v8f{};
        #pragma unroll
        for (int r = 0; r < 8; ++r) { mrow[rt][r] = -__builtin_inff(); lrow[rt][r] = 0.0f; }
    }

    const int nsteps = (t0 >> 5) + 1;
    prefetch_tile(kh + bT * H_DIM, vpk + bT * H_DIM, kbase, vbase, lane);

    for (int it = 0; it < nsteps; ++it) {
        const int s0  = it << 5;
        const int cur = it & 1;
        if (it + 1 < nsteps) {
            const long nb = (bT + s0 + 32) * H_DIM;
            prefetch_tile(kh + nb, vpk + nb,
                          kbase + (cur ^ 1) * 4096, vbase + (cur ^ 1) * 4096, lane);
            asm volatile("s_wait_asynccnt 16" ::: "memory");  // current tile resident
        } else {
            asm volatile("s_wait_asynccnt 0" ::: "memory");
        }

        // ---- scores from LDS K tile: 2 row tiles x 2 col tiles ----
        v8f sc[2][2];
        #pragma unroll
        for (int j = 0; j < 2; ++j) {
            v8f c0 = v8f{}, c1 = v8f{};
            #pragma unroll
            for (int hc = 0; hc < 2; ++hc) {
                const _Float16* kp = &kbuf[wslot][cur][(j * 16 + l15) * 64 + hc * 32 + (grp << 3)];
                U16h bk;
                bk.h[0] = *(const v8h*)(kp);
                bk.h[1] = *(const v8h*)(kp + 16);
                c0 = __builtin_amdgcn_wmma_f32_16x16x32_f16(
                    false, aq[0][hc], false, bk.v, (short)0, c0, false, false);
                c1 = __builtin_amdgcn_wmma_f32_16x16x32_f16(
                    false, aq[1][hc], false, bk.v, (short)0, c1, false, false);
            }
            sc[0][j] = c0; sc[1][j] = c1;
        }

        // ---- causal mask: only the diagonal step ----
        if (s0 == t0) {
            #pragma unroll
            for (int rt = 0; rt < 2; ++rt)
                #pragma unroll
                for (int j = 0; j < 2; ++j) {
                    int scol = s0 + j * 16 + l15;
                    #pragma unroll
                    for (int r = 0; r < 8; ++r)
                        if (scol > t0 + rt * 16 + r + (grp << 3))
                            sc[rt][j][r] = -__builtin_inff();
                }
        }

        // ---- online softmax; P written back into sc, then staged to LDS ----
        #pragma unroll
        for (int rt = 0; rt < 2; ++rt) {
            #pragma unroll
            for (int r = 0; r < 8; ++r) {
                float mx = fmaxf(sc[rt][0][r], sc[rt][1][r]);
                #pragma unroll
                for (int d = 1; d < 16; d <<= 1)
                    mx = fmaxf(mx, __shfl_xor(mx, d, 32));
                float nm    = fmaxf(mrow[rt][r], mx);
                float scale = __expf(mrow[rt][r] - nm);
                float p0 = __expf(sc[rt][0][r] - nm);
                float p1 = __expf(sc[rt][1][r] - nm);
                sc[rt][0][r] = p0; sc[rt][1][r] = p1;
                float sum = p0 + p1;
                #pragma unroll
                for (int d = 1; d < 16; d <<= 1)
                    sum += __shfl_xor(sum, d, 32);
                lrow[rt][r] = lrow[rt][r] * scale + sum;
                mrow[rt][r] = nm;
                #pragma unroll
                for (int n = 0; n < 4; ++n) o[rt][n][r] *= scale;
            }
            #pragma unroll
            for (int r = 0; r < 8; ++r) {
                int m = r + (grp << 3);
                plds[wslot][rt][m * 32 + l15]      = (_Float16)sc[rt][0][r];
                plds[wslot][rt][m * 32 + 16 + l15] = (_Float16)sc[rt][1][r];
            }
        }
        asm volatile("s_wait_dscnt 0" ::: "memory");   // same-wave LDS relayout fence

        v16h ap[2];
        #pragma unroll
        for (int rt = 0; rt < 2; ++rt) {
            const _Float16* pp = &plds[wslot][rt][l15 * 32 + (grp << 4)];
            U16h u;
            u.h[0] = *(const v8h*)(pp);
            u.h[1] = *(const v8h*)(pp + 8);
            ap[rt] = u.v;
        }

        // ---- O += P @ V from LDS packed-V tile ----
        #pragma unroll
        for (int n = 0; n < 4; ++n) {
            const _Float16* vp = &vbuf[wslot][cur][(n * 32 + lane) * 16];
            U16h bv;
            bv.h[0] = *(const v8h*)(vp);
            bv.h[1] = *(const v8h*)(vp + 8);
            o[0][n] = __builtin_amdgcn_wmma_f32_16x16x32_f16(
                false, ap[0], false, bv.v, (short)0, o[0][n], false, false);
            o[1][n] = __builtin_amdgcn_wmma_f32_16x16x32_f16(
                false, ap[1], false, bv.v, (short)0, o[1][n], false, false);
        }
    }

    // ---- epilogue ----
    #pragma unroll
    for (int rt = 0; rt < 2; ++rt)
        #pragma unroll
        for (int r = 0; r < 8; ++r) {
            float inv = 1.0f / lrow[rt][r];
            long row = bT + t0 + rt * 16 + r + (grp << 3);
            #pragma unroll
            for (int n = 0; n < 4; ++n)
                out[row * H_DIM + n * 16 + l15] = o[rt][n][r] * inv;
        }
}

// ---------------------------------------------------------------------------
extern "C" void kernel_launch(void* const* d_in, const int* in_sizes, int n_in,
                              void* d_out, int out_size, void* d_ws, size_t ws_size,
                              hipStream_t stream) {
    const float* x  = (const float*)d_in[0];
    const float* Wk = (const float*)d_in[1];
    const float* Wq = (const float*)d_in[2];
    const float* Wv = (const float*)d_in[3];
    float* out = (float*)d_out;

    const size_t per = (size_t)B_DIM * T_SEQ * H_DIM;   // 1M halves each
    _Float16* qh  = (_Float16*)d_ws;
    _Float16* kh  = qh + per;
    _Float16* vpk = kh + per;
    _Float16* pwq = vpk + per;                          // packed W: 3*C*H halves

    pack_w_kernel<<<48, 256, 0, stream>>>(Wq, Wk, Wv, pwq);
    qkv_proj_kernel<<<128, 256, 0, stream>>>(x, pwq, qh, kh, vpk);
    attn_kernel<<<128, 128, 0, stream>>>(qh, kh, vpk, out);
}
